// MultiHeadSelfAttention_48455821033775
// MI455X (gfx1250) — matrix operand order repaired and checked by reading, not compiled, not run on hardware
//
#include <hip/hip_runtime.h>

// ---------------- problem constants (from reference) ----------------
#define Bsz 4
#define Ssz 2048
#define Dsz 1024
#define Hsz 16
#define DKsz 64
#define BS (Bsz * Ssz)   // 8192 rows total

typedef unsigned short u16;
typedef __attribute__((ext_vector_type(16))) unsigned short v16u;
typedef __attribute__((ext_vector_type(16))) __bf16          v16bf;
typedef __attribute__((ext_vector_type(8)))  float           v8f;

// ---------------- helpers ----------------
__device__ __forceinline__ u16 f32_to_bf16(float f) {
  unsigned int u = __float_as_uint(f);
  u = u + 0x7FFFu + ((u >> 16) & 1u);   // round-to-nearest-even
  return (u16)(u >> 16);
}

union FragU {
  v16u v;
  uint4 q[2];
};

// A-fragment: 16x32 bf16 (MxK), row-major source, row stride ld (elements).
// Lanes 0-15: rows M=lane, K chunks {0..7,16..23}; lanes 16-31: K {8..15,24..31}.
__device__ __forceinline__ v16u load_a16(const u16* base, int ld, int row0,
                                         int k0, int lane) {
  int r  = row0 + (lane & 15);
  int kh = (lane >> 4) << 3;              // khalf * 8
  const u16* p = base + r * ld + k0;
  FragU f;
  f.q[0] = *(const uint4*)(p + kh);
  f.q[1] = *(const uint4*)(p + 16 + kh);
  return f.v;
}

// B-fragment: 32x16 bf16 (KxN) where B[k,n] = src[(col0+n)*ld + k0 + k].
// Lanes 0-15: K chunks {0..7,16..23}; lanes 16-31: K {8..15,24..31}.
__device__ __forceinline__ v16u load_bT16(const u16* base, int ld, int col0,
                                          int k0, int lane) {
  int c  = col0 + (lane & 15);
  int kh = (lane >> 4) << 3;              // khalf * 8
  const u16* p = base + c * ld + k0 + kh;
  FragU f;
  f.q[0] = *(const uint4*)p;
  f.q[1] = *(const uint4*)(p + 16);
  return f.v;
}

__device__ __forceinline__ v8f wmma_bf16(v16u a, v16u b, v8f c) {
  return __builtin_amdgcn_wmma_f32_16x16x32_bf16(
      false, __builtin_bit_cast(v16bf, a),
      false, __builtin_bit_cast(v16bf, b),
      (short)0, c, false, false);
}

// ---------------- kernel 0: f32 -> bf16 conversion (vectorized x8) --------
__global__ void cvt_f32_bf16(const float* __restrict__ src,
                             u16* __restrict__ dst, int n) {
  int i      = (blockIdx.x * blockDim.x + threadIdx.x) * 8;
  int stride = gridDim.x * blockDim.x * 8;
  for (; i < n; i += stride) {
    float4 f0 = *(const float4*)(src + i);
    float4 f1 = *(const float4*)(src + i + 4);
    uint4 o;
    o.x = (unsigned)f32_to_bf16(f0.x) | ((unsigned)f32_to_bf16(f0.y) << 16);
    o.y = (unsigned)f32_to_bf16(f0.z) | ((unsigned)f32_to_bf16(f0.w) << 16);
    o.z = (unsigned)f32_to_bf16(f1.x) | ((unsigned)f32_to_bf16(f1.y) << 16);
    o.w = (unsigned)f32_to_bf16(f1.z) | ((unsigned)f32_to_bf16(f1.w) << 16);
    *(uint4*)(dst + i) = o;
  }
}

// ---------------- pipelined GEMM core: C[32x64] += X[32xD] * W[64xD]^T ----
// Two independent fragment buffer sets (ping-pong, no copies) so the register
// allocator cannot merge them: loads for one set issue above the WMMAs of the
// other set and their latency hides under the matrix pipe.
__device__ __forceinline__ void gemm_tile_32x64(const u16* __restrict__ X,
                                                const u16* __restrict__ W,
                                                int mbase, int nbase, int lane,
                                                v8f c[2][4]) {
  v16u aA[2], bA[4], aB[2], bB[4];

  aA[0] = load_a16(X, Dsz, mbase, 0, lane);
  aA[1] = load_a16(X, Dsz, mbase + 16, 0, lane);
#pragma unroll
  for (int t = 0; t < 4; ++t)
    bA[t] = load_bT16(W, Dsz, nbase + t * 16, 0, lane);

  for (int k0 = 0; k0 < Dsz; k0 += 64) {
    // ---- issue loads for k0+32 into set B ----
    aB[0] = load_a16(X, Dsz, mbase, k0 + 32, lane);
    aB[1] = load_a16(X, Dsz, mbase + 16, k0 + 32, lane);
#pragma unroll
    for (int t = 0; t < 4; ++t)
      bB[t] = load_bT16(W, Dsz, nbase + t * 16, k0 + 32, lane);

    // ---- compute on set A (set-B loads in flight) ----
#pragma unroll
    for (int mt = 0; mt < 2; ++mt)
#pragma unroll
      for (int t = 0; t < 4; ++t)
        c[mt][t] = wmma_bf16(aA[mt], bA[t], c[mt][t]);

    // ---- issue loads for k0+64 into set A (dummy reload on last iter) ----
    int kn = (k0 + 64 < Dsz) ? k0 + 64 : 0;
    aA[0] = load_a16(X, Dsz, mbase, kn, lane);
    aA[1] = load_a16(X, Dsz, mbase + 16, kn, lane);
#pragma unroll
    for (int t = 0; t < 4; ++t)
      bA[t] = load_bT16(W, Dsz, nbase + t * 16, kn, lane);

    // ---- compute on set B (set-A loads in flight) ----
#pragma unroll
    for (int mt = 0; mt < 2; ++mt)
#pragma unroll
      for (int t = 0; t < 4; ++t)
        c[mt][t] = wmma_bf16(aB[mt], bB[t], c[mt][t]);
  }
}

// ---------------- kernel 1: fused QKV projection (y = x * W^T) ------------
// block = 128 (4 waves). Each wave: 32 rows x 64 cols.
// V is stored transposed: Vt[((b*H + h)*DK + dk)*S + s].
__global__ void qkv_proj_kernel(const u16* __restrict__ xb,
                                const u16* __restrict__ wq,
                                const u16* __restrict__ wk,
                                const u16* __restrict__ wv,
                                u16* __restrict__ Qb,
                                u16* __restrict__ Kb,
                                u16* __restrict__ Vtb) {
  int lane  = threadIdx.x & 31;
  int wave  = threadIdx.x >> 5;
  int mbase = (blockIdx.x * 4 + wave) * 32;
  int nbase = blockIdx.y * 64;
  int mat   = blockIdx.z;
  const u16* W = (mat == 0) ? wq : ((mat == 1) ? wk : wv);

  v8f c[2][4] = {};
  gemm_tile_32x64(xb, W, mbase, nbase, lane, c);

  int kh = lane >> 4, nn = lane & 15;
#pragma unroll
  for (int mt = 0; mt < 2; ++mt) {
#pragma unroll
    for (int t = 0; t < 4; ++t) {
#pragma unroll
      for (int r = 0; r < 8; ++r) {
        int row = mbase + mt * 16 + r + 8 * kh;   // global row in [0, B*S)
        int col = nbase + t * 16 + nn;            // [0, D)
        u16 hv = f32_to_bf16(c[mt][t][r]);
        if (mat == 0) {
          Qb[row * Dsz + col] = hv;
        } else if (mat == 1) {
          Kb[row * Dsz + col] = hv;
        } else {
          int b = row >> 11, s = row & (Ssz - 1);
          int hh = col >> 6, dk = col & 63;
          Vtb[((b * Hsz + hh) * DKsz + dk) * Ssz + s] = hv;
        }
      }
    }
  }
}

// ---------------- kernel 2: causal flash attention ----------------
// One wave per 16-query tile per (b,h). 64 keys per iteration (4 score tiles,
// 64-wide online softmax, two P fragments). Wave-uniform branches skip
// fully-masked tiles at the causal boundary.
__global__ void attn_kernel(const u16* __restrict__ Qb,
                            const u16* __restrict__ Kb,
                            const u16* __restrict__ Vtb,
                            u16* __restrict__ Ob) {
  __shared__ u16 Plds[4][16 * 64];       // per-wave P staging (bf16), 8 KB
  int lane = threadIdx.x & 31;
  int wave = threadIdx.x >> 5;
  int qt   = blockIdx.x * 4 + wave;
  int h    = blockIdx.y;
  int b    = blockIdx.z;
  int q0   = qt * 16;
  int qlast = q0 + 15;

  const u16* Qbase  = Qb + b * Ssz * Dsz + h * DKsz;      // row stride D
  const u16* Kbase  = Kb + b * Ssz * Dsz + h * DKsz;      // row stride D
  const u16* Vtbase = Vtb + (b * Hsz + h) * DKsz * Ssz;   // row stride S

  // Q fragments for K-dim d = 0..63 (two 16x32 A-frags), loaded once
  v16u a0 = load_a16(Qbase, Dsz, q0, 0, lane);
  v16u a1 = load_a16(Qbase, Dsz, q0, 32, lane);

  v8f acc[4] = {};
  float mrow[8], lrow[8];
#pragma unroll
  for (int r = 0; r < 8; ++r) { mrow[r] = -3.0e38f; lrow[r] = 0.0f; }

  const float scale = 0.125f;            // 1/sqrt(DK)
  int kh = lane >> 4, nn = lane & 15;
  u16* myP = &Plds[wave][0];

  // ---- prologue: K fragments for key block 0 (4 key tiles x 2 d-halves) ----
  v16u kf[4][2];
#pragma unroll
  for (int st = 0; st < 4; ++st) {
    kf[st][0] = load_bT16(Kbase, Dsz, st * 16, 0, lane);
    kf[st][1] = load_bT16(Kbase, Dsz, st * 16, 32, lane);
  }

  for (int kb = 0; kb <= q0; kb += 64) {
    // ---- S = Q*K^T, per 16-key tile (skip tiles fully above diagonal) ----
    v8f s[4] = {};
#pragma unroll
    for (int st = 0; st < 4; ++st) {
      if (kb + st * 16 <= qlast) {
        s[st] = wmma_bf16(a0, kf[st][0], s[st]);
        s[st] = wmma_bf16(a1, kf[st][1], s[st]);
      }
    }

    // ---- V fragments: issue now, consumed after the softmax ----
    bool half1 = (kb + 32 <= qlast);
    v16u vf0[4], vf1[4];
#pragma unroll
    for (int t = 0; t < 4; ++t)
      vf0[t] = load_bT16(Vtbase, Ssz, t * 16, kb, lane);
    if (half1) {
#pragma unroll
      for (int t = 0; t < 4; ++t)
        vf1[t] = load_bT16(Vtbase, Ssz, t * 16, kb + 32, lane);
    }

    // ---- prefetch next block's K fragments (hide under softmax/PV) ----
    int kbn = (kb + 64 <= q0) ? kb + 64 : kb;
    v16u kfn[4][2];
#pragma unroll
    for (int st = 0; st < 4; ++st) {
      kfn[st][0] = load_bT16(Kbase, Dsz, kbn + st * 16, 0, lane);
      kfn[st][1] = load_bT16(Kbase, Dsz, kbn + st * 16, 32, lane);
    }

    // ---- causal mask, scale, 64-wide running row-max ----
    float mnew[8];
#pragma unroll
    for (int r = 0; r < 8; ++r) {
      int row = q0 + r + 8 * kh;
      float mv = -3.0e38f;
#pragma unroll
      for (int st = 0; st < 4; ++st) {
        float v = s[st][r] * scale;
        if (kb + st * 16 + nn > row) v = -3.0e38f;
        s[st][r] = v;
        mv = fmaxf(mv, v);
      }
      mv = fmaxf(mv, __shfl_xor(mv, 1, 32));
      mv = fmaxf(mv, __shfl_xor(mv, 2, 32));
      mv = fmaxf(mv, __shfl_xor(mv, 4, 32));
      mv = fmaxf(mv, __shfl_xor(mv, 8, 32));   // stays within 16-lane half
      mnew[r] = fmaxf(mrow[r], mv);
    }

    // ---- P = exp(S - m), update l, rescale accumulators, stage P ----
#pragma unroll
    for (int r = 0; r < 8; ++r) {
      float alpha = __expf(mrow[r] - mnew[r]);
      float rs = 0.0f;
      int prow = r + 8 * kh;
#pragma unroll
      for (int st = 0; st < 4; ++st) {
        float p = __expf(s[st][r] - mnew[r]);
        rs += p;
        myP[prow * 64 + st * 16 + nn] = f32_to_bf16(p);
      }
      rs += __shfl_xor(rs, 1, 32);
      rs += __shfl_xor(rs, 2, 32);
      rs += __shfl_xor(rs, 4, 32);
      rs += __shfl_xor(rs, 8, 32);
      lrow[r] = lrow[r] * alpha + rs;
      mrow[r] = mnew[r];
#pragma unroll
      for (int t = 0; t < 4; ++t) acc[t][r] = acc[t][r] * alpha;
    }

    // ---- re-read P in A-fragment order (same wave; DScnt handled) ----
    FragU p0, p1;
    {
      const u16* pp = myP + (lane & 15) * 64 + (kh << 3);
      p0.q[0] = *(const uint4*)pp;
      p0.q[1] = *(const uint4*)(pp + 16);
      if (half1) {
        p1.q[0] = *(const uint4*)(pp + 32);
        p1.q[1] = *(const uint4*)(pp + 48);
      }
    }

    // ---- O += P * V ----
#pragma unroll
    for (int t = 0; t < 4; ++t)
      acc[t] = wmma_bf16(p0.v, vf0[t], acc[t]);
    if (half1) {
#pragma unroll
      for (int t = 0; t < 4; ++t)
        acc[t] = wmma_bf16(p1.v, vf1[t], acc[t]);
    }

#pragma unroll
    for (int st = 0; st < 4; ++st) {
      kf[st][0] = kfn[st][0];
      kf[st][1] = kfn[st][1];
    }
  }

  // ---- normalize and store attention output (bf16, [b,s,D] layout) ----
  u16* Obase = Ob + b * Ssz * Dsz + h * DKsz;
#pragma unroll
  for (int r = 0; r < 8; ++r) {
    float inv = 1.0f / lrow[r];
    int row = q0 + r + 8 * kh;
#pragma unroll
    for (int t = 0; t < 4; ++t)
      Obase[row * Dsz + t * 16 + nn] = f32_to_bf16(acc[t][r] * inv);
  }
}

// ---------------- kernel 3: output projection (f32 result) ----------------
__global__ void out_proj_kernel(const u16* __restrict__ Ob,
                                const u16* __restrict__ wo,
                                float* __restrict__ out) {
  int lane  = threadIdx.x & 31;
  int wave  = threadIdx.x >> 5;
  int mbase = (blockIdx.x * 4 + wave) * 32;
  int nbase = blockIdx.y * 64;

  v8f c[2][4] = {};
  gemm_tile_32x64(Ob, wo, mbase, nbase, lane, c);

  int kh = lane >> 4, nn = lane & 15;
#pragma unroll
  for (int mt = 0; mt < 2; ++mt)
#pragma unroll
    for (int t = 0; t < 4; ++t)
#pragma unroll
      for (int r = 0; r < 8; ++r)
        out[(mbase + mt * 16 + r + 8 * kh) * Dsz + nbase + t * 16 + nn] =
            c[mt][t][r];
}

// ---------------- host launcher ----------------
extern "C" void kernel_launch(void* const* d_in, const int* in_sizes, int n_in,
                              void* d_out, int out_size, void* d_ws,
                              size_t ws_size, hipStream_t stream) {
  const float* x  = (const float*)d_in[0];
  const float* wq = (const float*)d_in[1];
  const float* wk = (const float*)d_in[2];
  const float* wv = (const float*)d_in[3];
  const float* wo = (const float*)d_in[4];
  float* out = (float*)d_out;

  // workspace partition (bf16 elements)
  u16* xb  = (u16*)d_ws;
  u16* wqb = xb  + (size_t)BS * Dsz;
  u16* wkb = wqb + (size_t)Dsz * Dsz;
  u16* wvb = wkb + (size_t)Dsz * Dsz;
  u16* wob = wvb + (size_t)Dsz * Dsz;
  u16* Qb  = wob + (size_t)Dsz * Dsz;
  u16* Kb  = Qb  + (size_t)BS * Dsz;
  u16* Vtb = Kb  + (size_t)BS * Dsz;
  u16* Ob  = Vtb + (size_t)BS * Dsz;
  // total: ~92 MB of the workspace

  // 0) conversions to bf16
  cvt_f32_bf16<<<512, 256, 0, stream>>>(x,  xb,  BS * Dsz);
  cvt_f32_bf16<<<128, 256, 0, stream>>>(wq, wqb, Dsz * Dsz);
  cvt_f32_bf16<<<128, 256, 0, stream>>>(wk, wkb, Dsz * Dsz);
  cvt_f32_bf16<<<128, 256, 0, stream>>>(wv, wvb, Dsz * Dsz);
  cvt_f32_bf16<<<128, 256, 0, stream>>>(wo, wob, Dsz * Dsz);

  // 1) fused QKV projection (V stored transposed)
  {
    dim3 grid(BS / 128, Dsz / 64, 3);
    qkv_proj_kernel<<<grid, 128, 0, stream>>>(xb, wqb, wkb, wvb, Qb, Kb, Vtb);
  }

  // 2) causal flash attention
  {
    dim3 grid(Ssz / 64, Hsz, Bsz);   // 4 q-tiles (waves) per block
    attn_kernel<<<grid, 128, 0, stream>>>(Qb, Kb, Vtb, Ob);
  }

  // 3) output projection to f32
  {
    dim3 grid(BS / 128, Dsz / 64, 1);
    out_proj_kernel<<<grid, 128, 0, stream>>>(Ob, wob, out);
  }
}
